// StyleAttentionLayer_56934086476298
// MI455X (gfx1250) — matrix operand order repaired and checked by reading, not compiled, not run on hardware
//
#include <hip/hip_runtime.h>
#include <hip/hip_bf16.h>
#include <math.h>

typedef __attribute__((ext_vector_type(16))) __bf16 v16bf;
typedef __attribute__((ext_vector_type(8)))  __bf16 v8bf;
typedef __attribute__((ext_vector_type(8)))  float  v8f;

#define DIM_   1024
#define B_     8
#define T_     2048
#define S_     2048
#define H_     4
#define HD_    256
#define SCALE_ 0.03125f   /* DIM^-0.5 = 1/32 */

// K/V tile geometry in fragment-linear form: one s-tile (32 x 256 bf16) is
// 512 lane-chunks of 16 elements (32B). In LDS each 32B chunk is padded to 48B.
#define TILE_ELEMS   8192          /* 32*256 */
#define TILE_CHUNKS  512
#define LDS_TILE_B   (TILE_CHUNKS * 48)   /* 24576 B */
#define LDS_BUF_B    (2 * LDS_TILE_B)     /* K + V  = 49152 B */

static __device__ __forceinline__ v8f zero8() {
  v8f z;
#pragma unroll
  for (int i = 0; i < 8; ++i) z[i] = 0.0f;
  return z;
}

static __device__ __forceinline__ v16bf ld2(const __bf16* p0, const __bf16* p1) {
  v8bf a = *(const v8bf*)p0;
  v8bf b = *(const v8bf*)p1;
  return __builtin_shufflevector(a, b, 0,1,2,3,4,5,6,7,8,9,10,11,12,13,14,15);
}

static __device__ __forceinline__ v8f wmma_bf16(v16bf a, v16bf b, v8f c) {
  return __builtin_amdgcn_wmma_f32_16x16x32_bf16(false, a, false, b, (short)0, c,
                                                 false, false);
}

// CDNA5 async global->LDS copy, 16B per lane (tracked by ASYNCcnt).
static __device__ __forceinline__ void async_ld_b128(unsigned lds_off, unsigned goff,
                                                     const void* sbase) {
  asm volatile("global_load_async_to_lds_b128 %0, %1, %2"
               :: "v"(lds_off), "v"(goff), "s"(sbase) : "memory");
}
static __device__ __forceinline__ void wait_async8() {
  asm volatile("s_wait_asynccnt 0x8" ::: "memory");
}
static __device__ __forceinline__ void wait_async0() {
  asm volatile("s_wait_asynccnt 0x0" ::: "memory");
}

// ---------------------------------------------------------------------------
// Projection GEMM: C[M,1024] = act(A[M,1024] @ W[1024,1024]^T + bias)
// MODE 0: f32 in  -> bf16 out, natural [M,DIM] layout            (Q)
// MODE 1: f32 in  -> bf16 out, tanh, K fragment-swizzled layout  (K)
// MODE 3: f32 in  -> bf16 out, V fragment-swizzled layout        (V)
// MODE 2: bf16 in -> f32 out, * mask_t                           (final)
// ---------------------------------------------------------------------------
template<int MODE>
__global__ __launch_bounds__(256)
void proj_kernel(const void* __restrict__ Ain, const float* __restrict__ W,
                 const float* __restrict__ bias, const int* __restrict__ mask,
                 void* __restrict__ Cout) {
  __shared__ __align__(16) __bf16 As[128][40];
  __shared__ __align__(16) __bf16 Bs[128][40];

  const int tid  = threadIdx.x;
  const int lane = tid & 31;
  const int wave = tid >> 5;
  const int lo   = lane & 15;
  const int hi   = lane >> 4;
  const int wm   = wave & 1;
  const int wn   = wave >> 1;
  const int m0   = blockIdx.y * 128;
  const int n0   = blockIdx.x * 128;

  v8f acc[4][2];
#pragma unroll
  for (int i = 0; i < 4; ++i)
#pragma unroll
    for (int j = 0; j < 2; ++j) acc[i][j] = zero8();

  const int lrow = tid >> 1;
  const int lcol = (tid & 1) * 16;

  for (int k0 = 0; k0 < DIM_; k0 += 32) {
    __syncthreads();
    if (MODE == 2) {
      const __bf16* Ag = (const __bf16*)Ain + (size_t)(m0 + lrow) * DIM_ + k0 + lcol;
#pragma unroll
      for (int j = 0; j < 16; j += 8)
        *(v8bf*)&As[lrow][lcol + j] = *(const v8bf*)(Ag + j);
    } else {
      const float* Ag = (const float*)Ain + (size_t)(m0 + lrow) * DIM_ + k0 + lcol;
#pragma unroll
      for (int j = 0; j < 16; ++j) As[lrow][lcol + j] = (__bf16)Ag[j];
    }
    {
      const float* Wg = W + (size_t)(n0 + lrow) * DIM_ + k0 + lcol;
#pragma unroll
      for (int j = 0; j < 16; ++j) Bs[lrow][lcol + j] = (__bf16)Wg[j];
    }
    __syncthreads();

    v16bf af[4], bfr[2];
#pragma unroll
    for (int mt = 0; mt < 4; ++mt) {            // A operand: pair-packed layout
      const __bf16* r = &As[wm * 64 + mt * 16 + lo][0];
      af[mt] = ld2(r + 8 * hi, r + 8 * hi + 16);
    }
#pragma unroll
    for (int nt = 0; nt < 2; ++nt) {            // B operand: pair-packed layout
      const __bf16* r = &Bs[wn * 32 + nt * 16 + lo][0];
      bfr[nt] = ld2(r + 8 * hi, r + 8 * hi + 16);
    }
#pragma unroll
    for (int mt = 0; mt < 4; ++mt)
#pragma unroll
      for (int nt = 0; nt < 2; ++nt)
        acc[mt][nt] = wmma_bf16(af[mt], bfr[nt], acc[mt][nt]);
  }

#pragma unroll
  for (int mt = 0; mt < 4; ++mt) {
#pragma unroll
    for (int nt = 0; nt < 2; ++nt) {
      const int n  = n0 + wn * 32 + nt * 16 + lo;
      const float bv = bias[n];
#pragma unroll
      for (int r = 0; r < 8; ++r) {
        const int m = m0 + wm * 64 + mt * 16 + r + 8 * hi;
        float v = acc[mt][nt][r] + bv;
        if (MODE == 0) {
          ((__bf16*)Cout)[(size_t)m * DIM_ + n] = (__bf16)v;
        } else if (MODE == 2) {
          ((float*)Cout)[(size_t)m * DIM_ + n] = v * (float)mask[m];
        } else {
          const int bb    = m >> 11;            // m / S_
          const int s     = m & (S_ - 1);
          const int stile = s >> 5;
          const int srem  = s & 31;
          const int hh    = n >> 8;
          const int dh    = n & 255;
          size_t idx;
          if (MODE == 1) {                      // K: A-operand (pair-packed) chunks
            v = tanhf(v);
            const int st2  = srem >> 4, lo_s = srem & 15;
            const int c2   = dh >> 5,  drem = dh & 31;
            const int hi2  = (drem >> 3) & 1;
            const int ii   = drem - 8 * hi2 - 8 * (drem >> 4);
            const int lam  = hi2 * 16 + lo_s;
            idx = (((((size_t)(bb * H_ + hh) * 64 + stile) * 2 + st2) * 8 + c2) * 32
                   + lam) * 16 + ii;
          } else {                              // MODE 3, V: B-operand chunks
            const int hi_s = (srem >> 3) & 1, ii = srem - 8 * hi_s - 8 * (srem >> 4);
            const int j2   = dh >> 4,   lo_d = dh & 15;
            const int lam  = hi_s * 16 + lo_d;
            idx = ((((size_t)(bb * H_ + hh) * 64 + stile) * 16 + j2) * 32
                   + lam) * 16 + ii;
          }
          ((__bf16*)Cout)[idx] = (__bf16)v;
        }
      }
    }
  }
}

// ---------------------------------------------------------------------------
// Flash attention, one (head, batch, 128 T-rows) per workgroup, 8 waves x 16
// rows. K/V tiles arrive fragment-linear, double-buffered via
// global_load_async_to_lds_b128; every fragment read is ds_load_b128.
// ---------------------------------------------------------------------------
__global__ __launch_bounds__(256)
void attn_kernel(const __bf16* __restrict__ Q, const __bf16* __restrict__ Ksw,
                 const __bf16* __restrict__ Vsw, __bf16* __restrict__ O) {
  __shared__ __align__(16) unsigned char lds[2 * LDS_BUF_B];   // 96 KB

  const int tid  = threadIdx.x;
  const int lane = tid & 31;
  const int wave = tid >> 5;
  const int lo   = lane & 15;
  const int hi   = lane >> 4;

  const int hb = blockIdx.y;
  const int h  = hb >> 3;
  const int b  = hb & 7;
  const int t0 = blockIdx.x * 128 + wave * 16;

  const unsigned lds0 = (unsigned)(uintptr_t)&lds[0];
  const __bf16* Kbase = Ksw + (size_t)(b * H_ + h) * 64 * TILE_ELEMS;
  const __bf16* Vbase = Vsw + (size_t)(b * H_ + h) * 64 * TILE_ELEMS;

  // Q as WMMA-B operand: lane holds column t = t0 + lo, pair-packed d per half.
  v16bf qf[8];
  const __bf16* Qg = Q + ((size_t)b * T_ + t0 + lo) * DIM_ + h * HD_;
#pragma unroll
  for (int c = 0; c < 8; ++c) {
    const __bf16* p = Qg + 32 * c + 8 * hi;
    qf[c] = ld2(p, p + 16);
  }

  v8f o[16];
#pragma unroll
  for (int j = 0; j < 16; ++j) o[j] = zero8();
  float mx = -INFINITY, lsum = 0.0f;

  // issue one tile's async copy: each thread moves 64B of K and 64B of V
  auto stage = [&](int stile, int buf) {
    const __bf16* kg = Kbase + (size_t)stile * TILE_ELEMS;
    const __bf16* vg = Vbase + (size_t)stile * TILE_ELEMS;
    const unsigned base = lds0 + (unsigned)buf * LDS_BUF_B;
#pragma unroll
    for (int q2 = 0; q2 < 4; ++q2) {
      const unsigned g = (unsigned)tid * 64u + (unsigned)q2 * 16u;  // tile byte
      const unsigned l = (g >> 5) * 48u + (g & 31u);                // padded LDS
      async_ld_b128(base + l, g, kg);
      async_ld_b128(base + LDS_TILE_B + l, g, vg);
    }
  };

  stage(0, 0);
  const int NSTEP = S_ / 32;                    // 64
  for (int st = 0; st < NSTEP; ++st) {
    const int buf = st & 1;
    if (st + 1 < NSTEP) {                       // pipeline next tile
      stage(st + 1, buf ^ 1);
      wait_async8();                            // current tile's 8 xfers done
    } else {
      wait_async0();
    }
    __syncthreads();                            // all waves' data landed

    const unsigned char* kls = lds + (unsigned)buf * LDS_BUF_B;
    const unsigned char* vls = kls + LDS_TILE_B;

    // scores^T: A = K fragments (s rows), B = Q columns (t). Two 16-s tiles.
    v8f sc0 = zero8(), sc1 = zero8();
#pragma unroll
    for (int c = 0; c < 8; ++c) {
      const __bf16* p0 = (const __bf16*)(kls + (size_t)((0 * 8 + c) * 32 + lane) * 48);
      const __bf16* p1 = (const __bf16*)(kls + (size_t)((1 * 8 + c) * 32 + lane) * 48);
      sc0 = wmma_bf16(ld2(p0, p0 + 8), qf[c], sc0);
      sc1 = wmma_bf16(ld2(p1, p1 + 8), qf[c], sc1);
    }

    // flash softmax: per-t stats (t = lo); s spans regs x tiles x half-waves
    float pm = -INFINITY;
#pragma unroll
    for (int r = 0; r < 8; ++r) {
      sc0[r] *= SCALE_;
      sc1[r] *= SCALE_;
      pm = fmaxf(pm, fmaxf(sc0[r], sc1[r]));
    }
    pm = fmaxf(pm, __shfl_xor(pm, 16, 32));
    const float mnew = fmaxf(mx, pm);
    const float corr = __expf(mx - mnew);
    mx = mnew;

    float rsum = 0.0f;
    v16bf pf;                                   // P directly in WMMA-A layout
#pragma unroll
    for (int r = 0; r < 8; ++r) {
      const float p0 = __expf(sc0[r] - mnew);
      const float p1 = __expf(sc1[r] - mnew);
      rsum += p0 + p1;
      pf[r]     = (__bf16)p0;
      pf[r + 8] = (__bf16)p1;
    }
    rsum += __shfl_xor(rsum, 16, 32);
    lsum = lsum * corr + rsum;

    float rc[8];
#pragma unroll
    for (int r = 0; r < 8; ++r) rc[r] = __shfl(corr, r + 8 * hi, 32);

#pragma unroll
    for (int j = 0; j < 16; ++j) {
      v8f cacc = o[j];
#pragma unroll
      for (int r = 0; r < 8; ++r) cacc[r] *= rc[r];
      const __bf16* pv = (const __bf16*)(vls + (size_t)(j * 32 + lane) * 48);
      o[j] = wmma_bf16(pf, ld2(pv, pv + 8), cacc);
    }
    __syncthreads();                            // buf reusable for st+2's stage
  }

  float inv[8];
#pragma unroll
  for (int r = 0; r < 8; ++r) inv[r] = 1.0f / __shfl(lsum, r + 8 * hi, 32);
#pragma unroll
  for (int j = 0; j < 16; ++j) {
#pragma unroll
    for (int r = 0; r < 8; ++r) {
      const int t = t0 + r + 8 * hi;
      O[((size_t)b * T_ + t) * DIM_ + h * HD_ + 16 * j + lo] =
          (__bf16)(o[j][r] * inv[r]);
    }
  }
}

// ---------------------------------------------------------------------------
extern "C" void kernel_launch(void* const* d_in, const int* in_sizes, int n_in,
                              void* d_out, int out_size, void* d_ws, size_t ws_size,
                              hipStream_t stream) {
  const float* x    = (const float*)d_in[0];
  const float* keys = (const float*)d_in[1];
  const float* vals = (const float*)d_in[2];
  const int*   mask = (const int*)d_in[3];
  const float* Wq   = (const float*)d_in[4];
  const float* bq   = (const float*)d_in[5];
  const float* Wk   = (const float*)d_in[6];
  const float* bk   = (const float*)d_in[7];
  const float* Wv   = (const float*)d_in[8];
  const float* bv   = (const float*)d_in[9];
  const float* Wo   = (const float*)d_in[10];
  const float* bo   = (const float*)d_in[11];

  const size_t MN = (size_t)B_ * T_ * DIM_;
  __bf16* Qb = (__bf16*)d_ws;         // natural [B*T, DIM]
  __bf16* Kb = Qb + MN;               // fragment-swizzled
  __bf16* Vb = Kb + MN;               // fragment-swizzled
  __bf16* Ob = Vb + MN;               // natural [B*T, DIM]

  dim3 blk(256);
  dim3 gproj(DIM_ / 128, (B_ * T_) / 128);
  dim3 gattn(T_ / 128, H_ * B_);

  proj_kernel<0><<<gproj, blk, 0, stream>>>((const void*)x,    Wq, bq, nullptr, (void*)Qb);
  proj_kernel<1><<<gproj, blk, 0, stream>>>((const void*)keys, Wk, bk, nullptr, (void*)Kb);
  proj_kernel<3><<<gproj, blk, 0, stream>>>((const void*)vals, Wv, bv, nullptr, (void*)Vb);
  attn_kernel   <<<gattn, blk, 0, stream>>>(Qb, Kb, Vb, Ob);
  proj_kernel<2><<<gproj, blk, 0, stream>>>((const void*)Ob,   Wo, bo, mask,    d_out);
  (void)in_sizes; (void)n_in; (void)out_size; (void)ws_size;
}
